// SimpleRLLSTM_44152263803227
// MI455X (gfx1250) — matrix operand order repaired and checked by reading, not compiled, not run on hardware
//
#include <hip/hip_runtime.h>
#include <hip/hip_bf16.h>
#include <stdint.h>

// Problem dims
#define B_   256
#define T_   128
#define E_   256
#define H_   256
#define G_   1024        // 4*H
#define O_   1024
#define KFC_ 32768       // T*H

typedef __attribute__((ext_vector_type(16))) __bf16 bf16x16;
typedef __attribute__((ext_vector_type(8)))  float  f32x8;
typedef int v4i __attribute__((vector_size(16)));

#if defined(__has_builtin)
#if __has_builtin(__builtin_amdgcn_global_load_async_to_lds_b128) && \
    __has_builtin(__builtin_amdgcn_s_wait_asynccnt)
#define HAVE_ASYNC_LDS 1
#endif
#endif

#if HAVE_ASYNC_LDS
__device__ __forceinline__ void async_copy_b128(const void* gsrc, void* ldst) {
  __builtin_amdgcn_global_load_async_to_lds_b128(
      (__attribute__((address_space(1))) v4i*)gsrc,
      (__attribute__((address_space(3))) v4i*)ldst, 0, 0);
}
#endif

struct Pack32 { uint4 lo, hi; };

__device__ __forceinline__ bf16x16 pack_a(uint4 lo, uint4 hi) {
  Pack32 p{lo, hi};
  return __builtin_bit_cast(bf16x16, p);
}

__device__ __forceinline__ float sigmoidf_(float x) {
  return 1.0f / (1.0f + __expf(-x));
}
__device__ __forceinline__ float tanhf_(float x) {
  float e = __expf(-2.0f * fabsf(x));
  float r = (1.0f - e) / (1.0f + e);
  return copysignf(r, x);
}

// ---------------- fp32 -> bf16 conversion (8 elems/thread) ----------------
__global__ void cvt_f32_bf16(const float* __restrict__ src,
                             __bf16* __restrict__ dst, int n) {
  int i = (blockIdx.x * blockDim.x + threadIdx.x) * 8;
  if (i + 8 <= n) {
    float4 a = *(const float4*)(src + i);
    float4 b = *(const float4*)(src + i + 4);
    alignas(16) __bf16 o[8] = {(__bf16)a.x, (__bf16)a.y, (__bf16)a.z, (__bf16)a.w,
                               (__bf16)b.x, (__bf16)b.y, (__bf16)b.z, (__bf16)b.w};
    *(uint4*)(dst + i) = *(const uint4*)o;
  }
}

// ---------------- Stage 1: x_gates = x * w_ih^T + b_ih + b_hh -------------
__global__ void xgate_gemm(const __bf16* __restrict__ xbf,
                           const __bf16* __restrict__ wih,
                           const float* __restrict__ b_ih,
                           const float* __restrict__ b_hh,
                           float* __restrict__ xg) {
  const int lane = threadIdx.x & 31;
  const int wave = threadIdx.x >> 5;
  const int m0 = blockIdx.x * 16;
  const int n0 = (blockIdx.y * 8 + wave) * 16;
  const int lm = lane & 15;
  const int hi = lane >> 4;

  const __bf16* arow = xbf + (size_t)(m0 + lm) * E_;
  const __bf16* brow = wih + (size_t)(n0 + lm) * E_;

  f32x8 acc = {};
#pragma unroll
  for (int kt = 0; kt < E_ / 32; ++kt) {
    const int k0 = kt * 32;
    bf16x16 a = pack_a(*(const uint4*)(arow + k0 + hi * 16),
                       *(const uint4*)(arow + k0 + hi * 16 + 8));
    bf16x16 b = *(const bf16x16*)(brow + k0 + hi * 16);
    acc = __builtin_amdgcn_wmma_f32_16x16x32_bf16(false, a, false, b,
                                                  (short)0, acc, false, false);
  }
  const int col = n0 + lm;
  const float bias = b_ih[col] + b_hh[col];
#pragma unroll
  for (int v = 0; v < 8; ++v) {
    const int row = m0 + v + hi * 8;
    xg[(size_t)row * G_ + col] = acc[v] + bias;
  }
}

// ---------------- Stage 2: sequential LSTM scan ---------------------------
// Each workgroup owns 16 batch rows; h/c live in LDS; w_hh (bf16, 512KB)
// streams from L2 every step. 32 waves, each wave -> 2 N-tiles.
// The next timestep's 64KB x_gates slab is prefetched into LDS with the
// CDNA5 async global->LDS path (ASYNCcnt) so the serial-path elementwise
// phase never waits on global latency.
#define ROWS 16
#define HSTR 264     // h_lds row stride (bf16) : 528B -> conflict-spread
#define GSTR 1032    // gates row stride (f32)

__global__ __launch_bounds__(1024)
void lstm_scan(const __bf16* __restrict__ whh,
               const float* __restrict__ xg,
               __bf16* __restrict__ hs) {
  __shared__ __align__(16) __bf16 h_lds[ROWS * HSTR];
  __shared__ float c_lds[ROWS * H_];
  __shared__ float g_lds[ROWS * GSTR];
#if HAVE_ASYNC_LDS
  __shared__ __align__(16) float x_lds[ROWS * G_];   // 64KB staging for xg[t]
#endif

  const int tid  = threadIdx.x;
  const int lane = tid & 31;
  const int wave = tid >> 5;     // 0..31
  const int lm = lane & 15;
  const int hi = lane >> 4;
  const int r0 = blockIdx.x * ROWS;

#if HAVE_ASYNC_LDS
  // issue async prefetch of xg[:, t=0, :] for our batch rows (4x b128/thread)
  {
    const int row = tid >> 8;           // 4 rows per 1024 threads per pass
    const int c16 = (tid & 255) * 4;    // float col of this 16B chunk
#pragma unroll
    for (int k = 0; k < 4; ++k) {
      const int r = row + k * 4;
      const float* gsrc = xg + ((size_t)(r0 + r) * T_ + 0) * G_ + c16;
      float* ldst = x_lds + r * G_ + c16;
      async_copy_b128(gsrc, ldst);
    }
  }
#endif

  for (int i = tid; i < ROWS * HSTR; i += 1024) h_lds[i] = (__bf16)0.0f;
  for (int i = tid; i < ROWS * H_;   i += 1024) c_lds[i] = 0.0f;
  __syncthreads();

  for (int t = 0; t < T_; ++t) {
    // gates = h @ w_hh^T  (each wave: 2 output tiles of 16x16)
#pragma unroll
    for (int j = 0; j < 2; ++j) {
      const int n0 = (wave * 2 + j) * 16;
      const __bf16* brow = whh + (size_t)(n0 + lm) * H_;
      const __bf16* arow = h_lds + lm * HSTR;
      f32x8 acc = {};
#pragma unroll
      for (int kt = 0; kt < H_ / 32; ++kt) {
        const int k0 = kt * 32;
        bf16x16 a = pack_a(*(const uint4*)(arow + k0 + hi * 16),
                           *(const uint4*)(arow + k0 + hi * 16 + 8));
        bf16x16 b = *(const bf16x16*)(brow + k0 + hi * 16);
        acc = __builtin_amdgcn_wmma_f32_16x16x32_bf16(false, a, false, b,
                                                      (short)0, acc, false, false);
      }
#pragma unroll
      for (int v = 0; v < 8; ++v)
        g_lds[(v + hi * 8) * GSTR + n0 + lm] = acc[v];
    }

#if HAVE_ASYNC_LDS
    __builtin_amdgcn_s_wait_asynccnt(0);   // xg[t] staged in x_lds
#endif
    __syncthreads();

    // fused gate nonlinearity + state update (16 rows x 256 cols)
#pragma unroll
    for (int j = 0; j < 4; ++j) {
      const int idx = tid + j * 1024;
      const int row = idx >> 8;
      const int col = idx & 255;
#if HAVE_ASYNC_LDS
      const float* xgp = x_lds + row * G_ + col;
#else
      const float* xgp = xg + ((size_t)(r0 + row) * T_ + t) * G_ + col;
#endif
      float iv = sigmoidf_(g_lds[row * GSTR +        col] + xgp[0]);
      float fv = sigmoidf_(g_lds[row * GSTR + 256 + col] + xgp[256]);
      float gv =    tanhf_(g_lds[row * GSTR + 512 + col] + xgp[512]);
      float ov = sigmoidf_(g_lds[row * GSTR + 768 + col] + xgp[768]);
      float c = fv * c_lds[row * H_ + col] + iv * gv;
      c_lds[row * H_ + col] = c;
      float h = ov * tanhf_(c);
      __bf16 hb = (__bf16)h;
      h_lds[row * HSTR + col] = hb;
      hs[(size_t)(r0 + row) * KFC_ + t * H_ + col] = hb;   // [B, T*H]
    }
    __syncthreads();

#if HAVE_ASYNC_LDS
    // all threads are done reading x_lds for step t -> start staging t+1;
    // the copy overlaps the next step's WMMA phase.
    if (t + 1 < T_) {
      const int row = tid >> 8;
      const int c16 = (tid & 255) * 4;
#pragma unroll
      for (int k = 0; k < 4; ++k) {
        const int r = row + k * 4;
        const float* gsrc = xg + ((size_t)(r0 + r) * T_ + (t + 1)) * G_ + c16;
        float* ldst = x_lds + r * G_ + c16;
        async_copy_b128(gsrc, ldst);
      }
    }
#endif
  }
}

// ---------------- Stage 3: out = lstm_out @ w_fc^T + b_fc -----------------
__global__ void fc_gemm(const __bf16* __restrict__ hs,
                        const __bf16* __restrict__ wfc,
                        const float* __restrict__ b_fc,
                        float* __restrict__ out) {
  const int lane = threadIdx.x & 31;
  const int wave = threadIdx.x >> 5;
  const int m0 = blockIdx.x * 16;
  const int n0 = (blockIdx.y * 8 + wave) * 16;
  const int lm = lane & 15;
  const int hi = lane >> 4;

  const __bf16* arow = hs  + (size_t)(m0 + lm) * KFC_;
  const __bf16* brow = wfc + (size_t)(n0 + lm) * KFC_;

  f32x8 acc0 = {};
  f32x8 acc1 = {};
  for (int kt = 0; kt < KFC_ / 32; kt += 2) {
    __builtin_prefetch(brow + kt * 32 + 1024, 0, 1);  // global_prefetch_b8
    {
      const int k0 = kt * 32;
      bf16x16 a = pack_a(*(const uint4*)(arow + k0 + hi * 16),
                         *(const uint4*)(arow + k0 + hi * 16 + 8));
      bf16x16 b = *(const bf16x16*)(brow + k0 + hi * 16);
      acc0 = __builtin_amdgcn_wmma_f32_16x16x32_bf16(false, a, false, b,
                                                     (short)0, acc0, false, false);
    }
    {
      const int k0 = kt * 32 + 32;
      bf16x16 a = pack_a(*(const uint4*)(arow + k0 + hi * 16),
                         *(const uint4*)(arow + k0 + hi * 16 + 8));
      bf16x16 b = *(const bf16x16*)(brow + k0 + hi * 16);
      acc1 = __builtin_amdgcn_wmma_f32_16x16x32_bf16(false, a, false, b,
                                                     (short)0, acc1, false, false);
    }
  }
  const int col = n0 + lm;
  const float bias = b_fc[col];
#pragma unroll
  for (int v = 0; v < 8; ++v) {
    const int row = m0 + v + hi * 8;
    out[(size_t)row * O_ + col] = acc0[v] + acc1[v] + bias;
  }
}

// ---------------------------------------------------------------------------
extern "C" void kernel_launch(void* const* d_in, const int* in_sizes, int n_in,
                              void* d_out, int out_size, void* d_ws, size_t ws_size,
                              hipStream_t stream) {
  const float* x    = (const float*)d_in[0];
  const float* w_ih = (const float*)d_in[1];
  const float* w_hh = (const float*)d_in[2];
  const float* b_ih = (const float*)d_in[3];
  const float* b_hh = (const float*)d_in[4];
  const float* w_fc = (const float*)d_in[5];
  const float* b_fc = (const float*)d_in[6];
  float* out = (float*)d_out;

  // Workspace carve-up (256B aligned):
  char* ws = (char*)d_ws;
  size_t off = 0;
  auto carve = [&](size_t bytes) -> void* {
    void* p = ws + off;
    off += (bytes + 255) & ~(size_t)255;
    return p;
  };
  __bf16* xbf  = (__bf16*)carve((size_t)B_ * T_ * E_ * 2);   // 16.8 MB
  __bf16* wihb = (__bf16*)carve((size_t)G_ * E_ * 2);        // 0.5 MB
  __bf16* whhb = (__bf16*)carve((size_t)G_ * H_ * 2);        // 0.5 MB
  __bf16* wfcb = (__bf16*)carve((size_t)O_ * KFC_ * 2);      // 67 MB
  float*  xg   = (float*) carve((size_t)B_ * T_ * G_ * 4);   // 134 MB
  __bf16* hsb  = (__bf16*)carve((size_t)B_ * T_ * H_ * 2);   // 16.8 MB
  (void)ws_size; (void)in_sizes; (void)n_in; (void)out_size;

  // bf16 conversions
  {
    int n;
    n = B_ * T_ * E_; cvt_f32_bf16<<<n / 2048, 256, 0, stream>>>(x, xbf, n);
    n = G_ * E_;      cvt_f32_bf16<<<n / 2048, 256, 0, stream>>>(w_ih, wihb, n);
    n = G_ * H_;      cvt_f32_bf16<<<n / 2048, 256, 0, stream>>>(w_hh, whhb, n);
    n = O_ * KFC_;    cvt_f32_bf16<<<n / 2048, 256, 0, stream>>>(w_fc, wfcb, n);
  }

  // Stage 1: input projection for all timesteps
  {
    dim3 grid(B_ * T_ / 16, 8);   // 2048 x 8 blocks, 8 waves each
    xgate_gemm<<<grid, 256, 0, stream>>>(xbf, wihb, b_ih, b_hh, xg);
  }

  // Stage 2: sequential scan, batch partitioned across workgroups
  lstm_scan<<<B_ / ROWS, 1024, 0, stream>>>(whhb, xg, hsb);

  // Stage 3: FC head
  {
    dim3 grid(B_ / 16, O_ / 128);  // 16 x 8 blocks, 8 waves each
    fc_gemm<<<grid, 256, 0, stream>>>(hsb, wfcb, b_fc, out);
  }
}
